// LnLstm_59373627900386
// MI455X (gfx1250) — matrix operand order repaired and checked by reading, not compiled, not run on hardware
//
#include <hip/hip_runtime.h>
#include <stdint.h>

// ---------------------------------------------------------------------------
// LSTM (N=128, T=256, I=64, M=1024) for MI455X / gfx1250.
//
// gate_pre = mod_t @ Weff + h_{t-1} @ Wbot + beff, with
//   Weff = Wx @ Wtop  (exact: reference uses identity nonlinearity on x-proj)
//   beff = bx @ Wtop + b_gate
// One kernel per timestep; stream order provides the recurrence dependency.
// B matrices are stored FRAGMENT-MAJOR: for each (32-wide K block, 16-wide
// column tile), the 32 lanes' 8 bf16-pair words are contiguous -> each WMMA B
// operand is two global_load_b128 with immediate offsets.
// ---------------------------------------------------------------------------

#define N_  128
#define T_  256
#define I_  64
#define M_  1024
#define C4_ 4096   // 4 gates * M concatenated columns (i, f, g, o)
// fragment-major B geometry
#define CT_   (C4_ / 16)       // 256 column tiles
#define FWRDS 256              // words per fragment (32 lanes * 8)
#define KBLK  (CT_ * FWRDS)    // words per K-block slab = 65536 (256 KB)

typedef __attribute__((ext_vector_type(16))) __bf16 v16bf;
typedef __attribute__((ext_vector_type(8)))  float  v8f;

union FragAB {
  v16bf    v;
  uint32_t u[8];
};

__device__ __forceinline__ uint16_t f2bf(float f) {
  union { float f; uint32_t u; } x; x.f = f;
  uint32_t u = x.u;
  uint32_t r = u + 0x7FFFu + ((u >> 16) & 1u);   // round-to-nearest-even
  return (uint16_t)(r >> 16);
}
__device__ __forceinline__ uint32_t pack2(uint16_t lo, uint16_t hi) {
  return (uint32_t)lo | ((uint32_t)hi << 16);
}
__device__ __forceinline__ float sigm(float x) {
  return 1.0f / (1.0f + __expf(-x));
}
__device__ __forceinline__ const float* sel4(int g, const float* a, const float* b,
                                             const float* c, const float* d) {
  return g == 0 ? a : g == 1 ? b : g == 2 ? c : d;
}

// Word index of packed pair (k&~1, k|1) at column col, fragment-major layout.
// Lane half selects K run (contiguous order k = 16h + s, not ISA table order): half 0 -> K=kk..kk+15,
// half 1 -> K=kk+16..kk+31, VGPR v holds pair (2v, 2v+1) within the octet*2.
__device__ __forceinline__ size_t bfrag_idx(int k, int col) {
  int kb   = k >> 5;                 // K block
  int half = (k >> 4) & 1;
  int v    = (k >> 1) & 7;
  int lane = half * 16 + (col & 15);
  return ((size_t)(kb * CT_ + (col >> 4)) * 32 + lane) * 8 + v;
}

// --------------------------- prep kernels ----------------------------------

// Weff[k][col] = sum_j Wx[k][j] * Wtop_gate[j][m],  k<64, col = g*1024+m.
__global__ __launch_bounds__(256) void prep_weff_kernel(
    const float* __restrict__ Wx,
    const float* Wi, const float* Wf, const float* Wg, const float* Wo,
    uint16_t* __restrict__ weff_bf)
{
  int idx = blockIdx.x * 256 + threadIdx.x;        // 64*4096 threads
  int col = idx & (C4_ - 1);
  int k   = idx >> 12;
  int g = col >> 10, m = col & (M_ - 1);
  const float* W   = sel4(g, Wi, Wf, Wg, Wo);      // rows 0..M-1 = x part
  const float* wxr = Wx + (size_t)k * M_;
  float acc = 0.0f;
  for (int j = 0; j < M_; ++j) acc = fmaf(wxr[j], W[(size_t)j * M_ + m], acc);
  weff_bf[bfrag_idx(k, col) * 2 + (k & 1)] = f2bf(acc);
}

// beff[col] = b_gate[m] + sum_j bx[j] * Wtop_gate[j][m]
__global__ __launch_bounds__(256) void prep_beff_kernel(
    const float* bx,
    const float* bi, const float* bfp, const float* bg, const float* bo,
    const float* Wi, const float* Wf, const float* Wg, const float* Wo,
    float* __restrict__ beff)
{
  int col = blockIdx.x * 256 + threadIdx.x;        // 4096 threads
  int g = col >> 10, m = col & (M_ - 1);
  const float* W = sel4(g, Wi, Wf, Wg, Wo);
  const float* b = sel4(g, bi, bfp, bg, bo);
  float acc = b[m];
  for (int j = 0; j < M_; ++j) acc = fmaf(bx[j], W[(size_t)j * M_ + m], acc);
  beff[col] = acc;
}

// Wbot[k][col] = W_gate[1024+k][m] -> bf16, fragment-major.
__global__ __launch_bounds__(256) void prep_wbot_kernel(
    const float* Wi, const float* Wf, const float* Wg, const float* Wo,
    uint16_t* __restrict__ wbot_bf)
{
  int idx = blockIdx.x * 256 + threadIdx.x;        // 1024*4096 threads
  int col = idx & (C4_ - 1);
  int k   = idx >> 12;
  int g = col >> 10, m = col & (M_ - 1);
  const float* W = sel4(g, Wi, Wf, Wg, Wo);
  wbot_bf[bfrag_idx(k, col) * 2 + (k & 1)] = f2bf(W[(size_t)(M_ + k) * M_ + m]);
}

// modulation [N][T][I] fp32 -> bf16 pair words: word (n*T+t)*32 + (i>>1).
__global__ __launch_bounds__(256) void prep_mod_kernel(
    const float* __restrict__ mod, uint32_t* __restrict__ modbf)
{
  int idx = blockIdx.x * 256 + threadIdx.x;        // N*T*(I/2) = 1048576
  const float* s = mod + ((size_t)(idx >> 5) * I_) + ((idx & 31) << 1);
  modbf[idx] = pack2(f2bf(s[0]), f2bf(s[1]));
}

__global__ __launch_bounds__(256) void init_state_kernel(uint32_t* __restrict__ h0w,
                                                         float* __restrict__ c)
{
  int idx = blockIdx.x * 256 + threadIdx.x;        // 131072 threads
  c[idx] = 0.0f;
  if (idx < (N_ * M_ / 2)) h0w[idx] = 0u;          // h0 as packed bf16 pairs
}

// --------------------------- per-timestep kernel ---------------------------
//
// Grid (16, 2), block 256 (8 wave32): waves = 2 row-groups x 4 col-groups.
// Each wave: 32 rows (2 A frags) x 16 cols x 4 gates = 8 f32 accumulators;
// every B fragment feeds two WMMAs. Elementwise LSTM update done in-register.
__global__ __launch_bounds__(256) void lstm_step_kernel(
    const uint32_t* __restrict__ modbf,  // [N][T][32] bf16-pair words
    const uint32_t* __restrict__ weff,   // fragment-major, K=64  (512 KB)
    const uint32_t* __restrict__ wbot,   // fragment-major, K=1024 (8 MB)
    const float*    __restrict__ beff,   // [4096]
    const uint32_t* __restrict__ h_rd,   // [128][512] bf16-pair words (h_{t-1})
    uint16_t*       __restrict__ h_wr,   // [128][1024] bf16 (h_t)
    float*          __restrict__ c,      // [128][1024] fp32 cell state
    float*          __restrict__ out,    // [128][256][1024] fp32
    int t)
{
  const int tid  = threadIdx.x;
  const int lane = tid & 31;
  const int wave = tid >> 5;
  const int l15  = lane & 15;
  const int half = lane >> 4;
  const int rowbase = blockIdx.y * 64 + (wave >> 2) * 32;   // 32-row group
  const int mbase   = blockIdx.x * 64 + (wave & 3) * 16;    // 16-col group
  const int r0 = rowbase + l15;                             // A frag 0 row
  const int r1 = r0 + 16;                                   // A frag 1 row
  const int m  = mbase + l15;                               // C column
  const int ct = mbase >> 4;                                // column tile

  // Loop-invariant per-gate B lane pointers; all K blocks reachable via
  // immediate offsets (<= ~7.75 MB < 24-bit signed IOFFSET).
  const uint32_t* wbg[4];
#pragma unroll
  for (int g = 0; g < 4; ++g)
    wbg[g] = wbot + ((size_t)(ct + g * 64) * 32 + lane) * 8;

  // bias folded into accumulator init (all elements of a lane share column m)
  v8f acc[2][4];
#pragma unroll
  for (int g = 0; g < 4; ++g) {
    float b = beff[g * M_ + m];
    v8f bv = {b, b, b, b, b, b, b, b};
    acc[0][g] = bv;
    acc[1][g] = bv;
  }

  // ---- x part: mod_t @ Weff, K = 64 ----
  {
    const uint32_t* m0 = modbf + ((size_t)r0 * T_ + t) * (I_ / 2);
    const uint32_t* m1 = modbf + ((size_t)r1 * T_ + t) * (I_ / 2);
#pragma unroll
    for (int kk = 0; kk < I_; kk += 32) {
      FragAB a0, a1;
#pragma unroll
      for (int v = 0; v < 8; ++v) {
        int kp = (kk >> 1) + half * 8 + v;
        a0.u[v] = m0[kp];
        a1.u[v] = m1[kp];
      }
#pragma unroll
      for (int g = 0; g < 4; ++g) {
        const uint32_t* wp = weff + ((size_t)((kk >> 5) * CT_ + ct + g * 64) * 32 + lane) * 8;
        FragAB b;
#pragma unroll
        for (int v = 0; v < 8; ++v) b.u[v] = wp[v];     // 2x b128 per lane
        acc[0][g] = __builtin_amdgcn_wmma_f32_16x16x32_bf16(
            false, a0.v, false, b.v, (short)0, acc[0][g], false, false);
        acc[1][g] = __builtin_amdgcn_wmma_f32_16x16x32_bf16(
            false, a1.v, false, b.v, (short)0, acc[1][g], false, false);
      }
    }
  }

  // ---- recurrent part: h_{t-1} @ Wbot, K = 1024 ----
  {
    const uint32_t* h0 = h_rd + (size_t)r0 * (M_ / 2);
    const uint32_t* h1 = h_rd + (size_t)r1 * (M_ / 2);
#pragma unroll 2
    for (int kk = 0; kk < M_; kk += 32) {
      FragAB a0, a1;
#pragma unroll
      for (int v = 0; v < 8; ++v) {
        int kp = (kk >> 1) + half * 8 + v;
        a0.u[v] = h0[kp];
        a1.u[v] = h1[kp];
      }
      const size_t kslab = (size_t)(kk >> 5) * KBLK;     // immediate-foldable
#pragma unroll
      for (int g = 0; g < 4; ++g) {
        FragAB b;
#pragma unroll
        for (int v = 0; v < 8; ++v) b.u[v] = wbg[g][kslab + v];  // 2x b128
        acc[0][g] = __builtin_amdgcn_wmma_f32_16x16x32_bf16(
            false, a0.v, false, b.v, (short)0, acc[0][g], false, false);
        acc[1][g] = __builtin_amdgcn_wmma_f32_16x16x32_bf16(
            false, a1.v, false, b.v, (short)0, acc[1][g], false, false);
      }
    }
  }

  // ---- elementwise LSTM update, in-register (C layout: VGPR e -> row) ----
#pragma unroll
  for (int rg = 0; rg < 2; ++rg) {
#pragma unroll
    for (int e = 0; e < 8; ++e) {
      int n = rowbase + rg * 16 + half * 8 + e;
      float iv = sigm(acc[rg][0][e]);
      float fv = sigm(acc[rg][1][e]);
      float gv = tanhf(acc[rg][2][e]);
      float ov = sigm(acc[rg][3][e]);
      size_t idx = (size_t)n * M_ + m;
      float cn = fmaf(fv, c[idx], iv * gv);
      c[idx] = cn;
      float hv = ov * tanhf(cn);
      h_wr[idx] = f2bf(hv);
      out[(size_t)n * (T_ * M_) + (size_t)t * M_ + m] = hv;
    }
  }
}

// --------------------------- host launcher ---------------------------------

extern "C" void kernel_launch(void* const* d_in, const int* in_sizes, int n_in,
                              void* d_out, int out_size, void* d_ws, size_t ws_size,
                              hipStream_t stream) {
  const float* mod = (const float*)d_in[0];
  const float* Wx  = (const float*)d_in[1];
  const float* bx  = (const float*)d_in[2];
  const float* Wi  = (const float*)d_in[3];
  const float* bi  = (const float*)d_in[4];
  const float* Wf  = (const float*)d_in[5];
  const float* bfp = (const float*)d_in[6];
  const float* Wg  = (const float*)d_in[7];
  const float* bg  = (const float*)d_in[8];
  const float* Wo  = (const float*)d_in[9];
  const float* bo  = (const float*)d_in[10];
  float* out = (float*)d_out;

  char* ws = (char*)d_ws;
  size_t off = 0;
  uint16_t* weff_bf = (uint16_t*)(ws + off); off += (size_t)I_ * C4_ * 2;       // 512 KB
  float*    beff    = (float*)   (ws + off); off += (size_t)C4_ * 4;            // 16 KB
  uint16_t* wbot_bf = (uint16_t*)(ws + off); off += (size_t)M_ * C4_ * 2;       // 8 MB
  uint32_t* modbf   = (uint32_t*)(ws + off); off += (size_t)N_ * T_ * I_ * 2;   // 4 MB
  uint16_t* h0      = (uint16_t*)(ws + off); off += (size_t)N_ * M_ * 2;        // 256 KB
  uint16_t* h1      = (uint16_t*)(ws + off); off += (size_t)N_ * M_ * 2;        // 256 KB
  float*    c       = (float*)   (ws + off); off += (size_t)N_ * M_ * 4;        // 512 KB
  if (ws_size < off) return;  // ~13.5 MB needed

  prep_weff_kernel<<<(I_ * C4_) / 256, 256, 0, stream>>>(Wx, Wi, Wf, Wg, Wo, weff_bf);
  prep_beff_kernel<<<C4_ / 256, 256, 0, stream>>>(bx, bi, bfp, bg, bo,
                                                  Wi, Wf, Wg, Wo, beff);
  prep_wbot_kernel<<<(M_ * C4_) / 256, 256, 0, stream>>>(Wi, Wf, Wg, Wo, wbot_bf);
  prep_mod_kernel<<<(N_ * T_ * (I_ / 2)) / 256, 256, 0, stream>>>(mod, modbf);
  init_state_kernel<<<(N_ * M_) / 256, 256, 0, stream>>>((uint32_t*)h0, c);

  dim3 grid(M_ / 64, N_ / 64);   // (16, 2)
  dim3 block(256);
  for (int t = 0; t < T_; ++t) {
    const uint16_t* hr = (t & 1) ? h1 : h0;   // double-buffer h across steps
    uint16_t*       hw = (t & 1) ? h0 : h1;
    lstm_step_kernel<<<grid, block, 0, stream>>>(
        modbf, (const uint32_t*)weff_bf, (const uint32_t*)wbot_bf, beff,
        (const uint32_t*)hr, hw, c, out, t);
  }
}
